// BufferClassifier_48206712930927
// MI455X (gfx1250) — matrix operand order repaired and checked by reading, not compiled, not run on hardware
//
#include <hip/hip_runtime.h>

#define B_ROWS   32768
#define HID      4096
#define FIN      140
#define FPAD     160
#define MROWS    32            // rows per block (two 16-row WMMA groups)
#define LDSTRIDE 4104          // 4096 + 8 bf16 pad -> row m starts at bank 4*m (no conflicts)

typedef __attribute__((ext_vector_type(16))) __bf16 v16bf;
typedef __attribute__((ext_vector_type(8)))  float  v8f;

union Frag32 { uint4 u[2]; v16bf v; };

__device__ __forceinline__ unsigned short to_bf16(float f) {
    __bf16 h = (__bf16)f;                    // native v_cvt on gfx1250
    return __builtin_bit_cast(unsigned short, h);
}

__device__ __forceinline__ v8f wmma_bf16(v16bf a, v16bf b, v8f c) {
    return __builtin_amdgcn_wmma_f32_16x16x32_bf16(
        /*neg_a=*/false, a, /*neg_b=*/false, b,
        /*c_mod=*/(short)0, c, /*reuse_a=*/false, /*reuse_b=*/false);
}

// B fragment: two 8-bf16 chunks at +0 and +16 elements (lane base = &W[n][kbase + 8*half])
__device__ __forceinline__ v16bf load_b_frag(const unsigned short* p) {
    Frag32 f;
    const uint4* q = (const uint4*)p;
    f.u[0] = q[0];
    f.u[1] = q[2];
    return f.v;
}

// A fragment: two 8-bf16 chunks at +0 and +16 elements (lane base = row + kbase + 8*half)
__device__ __forceinline__ v16bf load_a_frag(const unsigned short* p) {
    Frag32 f;
    const uint4* q = (const uint4*)p;
    f.u[0] = q[0];
    f.u[1] = q[2];
    return f.v;
}

// ---------------------------------------------------------------- prep kernels
__global__ void cvt_whidden_kernel(const float* __restrict__ w, unsigned short* __restrict__ o) {
    int idx = blockIdx.x * blockDim.x + threadIdx.x;
    if (idx >= HID * FPAD) return;
    int n = idx / FPAD, k = idx % FPAD;
    float v = (k < FIN) ? w[n * FIN + k] : 0.0f;
    o[idx] = to_bf16(v);
}

__global__ void cvt_wff_kernel(const float* __restrict__ w, unsigned short* __restrict__ o) {
    int idx = blockIdx.x * blockDim.x + threadIdx.x;
    if (idx >= HID * HID) return;
    o[idx] = to_bf16(w[idx]);
}

// ---------------------------------------------------------------- features
__device__ void rnn10(const float* x, const float* __restrict__ wih,
                      const float* __restrict__ whh, const float* __restrict__ bih,
                      const float* __restrict__ bhh, float* out) {
    float h[10];
#pragma unroll
    for (int i = 0; i < 10; ++i) h[i] = 0.0f;
#pragma unroll 1
    for (int t = 0; t < 10; ++t) {
        float xt = x[t];
        float hn[10];
#pragma unroll
        for (int i = 0; i < 10; ++i) {
            float s = xt * wih[i] + bih[i] + bhh[i];
#pragma unroll
            for (int j = 0; j < 10; ++j) s += whh[i * 10 + j] * h[j];
            hn[i] = tanhf(s);
        }
#pragma unroll
        for (int i = 0; i < 10; ++i) h[i] = hn[i];
    }
#pragma unroll
    for (int i = 0; i < 10; ++i) out[i] = h[i];
}

__global__ void feat_kernel(
    const float* __restrict__ data,
    const float* __restrict__ emb_client, const float* __restrict__ emb_lastreq,
    const float* __restrict__ w_req, const float* __restrict__ b_req,
    const float* __restrict__ w_seq, const float* __restrict__ b_seq,
    const float* __restrict__ w_tac, const float* __restrict__ b_tac,
    const float* __restrict__ w_tcl, const float* __restrict__ b_tcl,
    const float* __restrict__ w_tl,  const float* __restrict__ b_tl,
    const float* __restrict__ w_mem, const float* __restrict__ b_mem,
    const float* __restrict__ w_cpu, const float* __restrict__ b_cpu,
    const float* __restrict__ pw_wih, const float* __restrict__ pw_whh,
    const float* __restrict__ pw_bih, const float* __restrict__ pw_bhh,
    const float* __restrict__ h_wih,  const float* __restrict__ h_whh,
    const float* __restrict__ h_bih,  const float* __restrict__ h_bhh,
    unsigned short* __restrict__ featbf)
{
    int row = blockIdx.x * blockDim.x + threadIdx.x;
    if (row >= B_ROWS) return;
    float d[70];
    const float* src = data + (long)row * 70;
#pragma unroll
    for (int i = 0; i < 70; ++i) d[i] = src[i];

    float f[FIN];
    int c1 = (int)d[1]; c1 = c1 < 0 ? 0 : (c1 > 2 ? 2 : c1);
    int c2 = (int)d[2]; c2 = c2 < 0 ? 0 : (c2 > 3 ? 3 : c2);
#pragma unroll
    for (int i = 0; i < 10; ++i) {
        f[i]      = emb_client[c1 * 10 + i];
        f[10 + i] = emb_lastreq[c2 * 10 + i];
        f[20 + i] = d[3] * w_req[i] + b_req[i];
        f[30 + i] = d[4] * w_seq[i] + b_seq[i];
        f[40 + i] = d[5] * w_tac[i] + b_tac[i];
        f[50 + i] = d[6] * w_tcl[i] + b_tcl[i];
        f[60 + i] = d[7] * w_tl[i]  + b_tl[i];
        float s = b_mem[i];
#pragma unroll
        for (int j = 0; j < 11; ++j) s += d[8 + j] * w_mem[i * 11 + j];
        f[70 + i] = s;
        f[80 + i] = d[19] * w_cpu[i] + b_cpu[i];
    }
    rnn10(&d[20], pw_wih, pw_whh, pw_bih, pw_bhh, &f[90]);
    rnn10(&d[30], h_wih,  h_whh,  h_bih,  h_bhh,  &f[100]);
    rnn10(&d[40], h_wih,  h_whh,  h_bih,  h_bhh,  &f[110]);
    rnn10(&d[50], h_wih,  h_whh,  h_bih,  h_bhh,  &f[120]);
    rnn10(&d[60], h_wih,  h_whh,  h_bih,  h_bhh,  &f[130]);

    unsigned short* o = featbf + (long)row * FPAD;
#pragma unroll
    for (int k = 0; k < FIN; ++k) o[k] = to_bf16(f[k]);
#pragma unroll
    for (int k = FIN; k < FPAD; ++k) o[k] = 0;
}

// ---------------------------------------------------------------- fused MLP
// 1 block = 32 rows (two 16-row WMMA groups).
// GEMM1 -> relu -> bf16 hid tile in LDS (padded stride) -> GEMM2 (2x2 outer product) -> logits.
__global__ void __launch_bounds__(256)
fused_mlp_kernel(const unsigned short* __restrict__ featbf,
                 const unsigned short* __restrict__ whbf,
                 const unsigned short* __restrict__ wfbf,
                 const float* __restrict__ b_hidden,
                 const float* __restrict__ b_ff,
                 float* __restrict__ out)
{
    extern __shared__ unsigned short sh[];   // MROWS x LDSTRIDE bf16 (~257 KB of 320 KB/WGP)
    const int tid  = threadIdx.x;
    const int wave = tid >> 5;
    const int lane = tid & 31;
    const int m    = lane & 15;
    const int half = lane >> 4;
    const long rbase = (long)blockIdx.x * MROWS;

    // ---- GEMM1: hid = relu(feat @ w_hidden.T + b_hidden), K = 160 (zero-padded)
    v16bf aF[2][5];
#pragma unroll
    for (int g = 0; g < 2; ++g) {
        const unsigned short* fr = featbf + (rbase + g * 16 + m) * FPAD + 8 * half;
#pragma unroll
        for (int kk = 0; kk < 5; ++kk) aF[g][kk] = load_a_frag(fr + kk * 32);
    }
#pragma unroll 1
    for (int g = 0; g < 2; ++g) {
#pragma unroll 1
        for (int i = 0; i < 32; ++i) {           // 8 waves x 32 tiles = 4096 cols
            const int n = (wave * 32 + i) * 16 + m;
            v8f acc = {};
            const unsigned short* wr = whbf + (long)n * FPAD + 8 * half;
#pragma unroll
            for (int kk = 0; kk < 5; ++kk)
                acc = wmma_bf16(aF[g][kk], load_b_frag(wr + kk * 32), acc);
            const float bias = b_hidden[n];
#pragma unroll
            for (int r = 0; r < 8; ++r)
                sh[(g * 16 + r + 8 * half) * LDSTRIDE + n] = to_bf16(fmaxf(acc[r] + bias, 0.0f));
        }
    }
    __syncthreads();

    // ---- GEMM2: logits = hid @ w_ff.T + b_ff, K = 4096 (w_ff bf16 stays L2-resident)
    const unsigned short* a0row = sh + (m)      * LDSTRIDE + 8 * half;   // rows 0..15
    const unsigned short* a1row = sh + (16 + m) * LDSTRIDE + 8 * half;   // rows 16..31
#pragma unroll 1
    for (int tg = 0; tg < 16; ++tg) {            // 2 n-tiles x 2 m-groups per step
        const int n0 = (wave * 32 + tg * 2) * 16 + m;
        const int n1 = n0 + 16;
        v8f acc00 = {}, acc01 = {}, acc10 = {}, acc11 = {};
        const unsigned short* w0 = wfbf + (long)n0 * HID + 8 * half;
        const unsigned short* w1 = wfbf + (long)n1 * HID + 8 * half;
#pragma unroll 2
        for (int kk = 0; kk < 128; ++kk) {
            const int kb = kk * 32;
            v16bf a0 = load_a_frag(a0row + kb);
            v16bf a1 = load_a_frag(a1row + kb);
            v16bf b0 = load_b_frag(w0 + kb);
            v16bf b1 = load_b_frag(w1 + kb);
            acc00 = wmma_bf16(a0, b0, acc00);
            acc01 = wmma_bf16(a0, b1, acc01);
            acc10 = wmma_bf16(a1, b0, acc10);
            acc11 = wmma_bf16(a1, b1, acc11);
        }
        const float bias0 = b_ff[n0];
        const float bias1 = b_ff[n1];
#pragma unroll
        for (int r = 0; r < 8; ++r) {
            const long row0 = rbase + r + 8 * half;        // group 0
            const long row1 = row0 + 16;                   // group 1
            out[row0 * HID + n0] = acc00[r] + bias0;
            out[row0 * HID + n1] = acc01[r] + bias1;
            out[row1 * HID + n0] = acc10[r] + bias0;
            out[row1 * HID + n1] = acc11[r] + bias1;
        }
    }
}

// ---------------------------------------------------------------- softmax
__global__ void __launch_bounds__(256) softmax_kernel(float* __restrict__ out) {
    __shared__ float red[256];
    float* p = out + (long)blockIdx.x * HID;
    const int tid = threadIdx.x;
    float mx = -3.4e38f;
    for (int c = tid; c < HID; c += 256) mx = fmaxf(mx, p[c]);
    red[tid] = mx; __syncthreads();
    for (int s = 128; s > 0; s >>= 1) {
        if (tid < s) red[tid] = fmaxf(red[tid], red[tid + s]);
        __syncthreads();
    }
    mx = red[0]; __syncthreads();
    float sum = 0.0f;
    for (int c = tid; c < HID; c += 256) {
        float e = __expf(p[c] - mx);
        p[c] = e;
        sum += e;
    }
    red[tid] = sum; __syncthreads();
    for (int s = 128; s > 0; s >>= 1) {
        if (tid < s) red[tid] += red[tid + s];
        __syncthreads();
    }
    float inv = 1.0f / red[0];
    __syncthreads();
    for (int c = tid; c < HID; c += 256) p[c] *= inv;
}

// ---------------------------------------------------------------- launch
extern "C" void kernel_launch(void* const* d_in, const int* in_sizes, int n_in,
                              void* d_out, int out_size, void* d_ws, size_t ws_size,
                              hipStream_t stream) {
    const float* data        = (const float*)d_in[0];
    const float* emb_client  = (const float*)d_in[1];
    const float* emb_lastreq = (const float*)d_in[2];
    const float* w_req = (const float*)d_in[3];  const float* b_req = (const float*)d_in[4];
    const float* w_seq = (const float*)d_in[5];  const float* b_seq = (const float*)d_in[6];
    const float* w_tac = (const float*)d_in[7];  const float* b_tac = (const float*)d_in[8];
    const float* w_tcl = (const float*)d_in[9];  const float* b_tcl = (const float*)d_in[10];
    const float* w_tl  = (const float*)d_in[11]; const float* b_tl  = (const float*)d_in[12];
    const float* w_mem = (const float*)d_in[13]; const float* b_mem = (const float*)d_in[14];
    const float* w_cpu = (const float*)d_in[15]; const float* b_cpu = (const float*)d_in[16];
    const float* pw_wih = (const float*)d_in[17]; const float* pw_whh = (const float*)d_in[18];
    const float* pw_bih = (const float*)d_in[19]; const float* pw_bhh = (const float*)d_in[20];
    const float* h_wih  = (const float*)d_in[21]; const float* h_whh  = (const float*)d_in[22];
    const float* h_bih  = (const float*)d_in[23]; const float* h_bhh  = (const float*)d_in[24];
    const float* w_hidden = (const float*)d_in[25]; const float* b_hidden = (const float*)d_in[26];
    const float* w_ff     = (const float*)d_in[27]; const float* b_ff     = (const float*)d_in[28];
    float* out = (float*)d_out;

    // workspace layout (bf16 staging)
    const size_t FEAT_BYTES = (size_t)B_ROWS * FPAD * 2;   // 10.0 MB
    const size_t WH_BYTES   = (size_t)HID * FPAD * 2;      //  1.25 MB
    unsigned short* featbf = (unsigned short*)d_ws;
    unsigned short* whbf   = (unsigned short*)((char*)d_ws + FEAT_BYTES);
    unsigned short* wfbf   = (unsigned short*)((char*)d_ws + FEAT_BYTES + WH_BYTES);

    cvt_whidden_kernel<<<(HID * FPAD + 255) / 256, 256, 0, stream>>>(w_hidden, whbf);
    cvt_wff_kernel<<<(HID * HID + 255) / 256, 256, 0, stream>>>(w_ff, wfbf);

    feat_kernel<<<(B_ROWS + 255) / 256, 256, 0, stream>>>(
        data, emb_client, emb_lastreq,
        w_req, b_req, w_seq, b_seq, w_tac, b_tac, w_tcl, b_tcl, w_tl, b_tl,
        w_mem, b_mem, w_cpu, b_cpu,
        pw_wih, pw_whh, pw_bih, pw_bhh, h_wih, h_whh, h_bih, h_bhh, featbf);

    const int LDS_BYTES = MROWS * LDSTRIDE * 2;   // ~257 KB hid tile (<= 320 KB/WGP)
    (void)hipFuncSetAttribute((const void*)fused_mlp_kernel,
                              hipFuncAttributeMaxDynamicSharedMemorySize, LDS_BYTES);
    fused_mlp_kernel<<<B_ROWS / MROWS, 256, LDS_BYTES, stream>>>(
        featbf, whbf, wfbf, b_hidden, b_ff, out);

    softmax_kernel<<<B_ROWS, 256, 0, stream>>>(out);
}
